// AdaptiveSubgraphLayer_35940286333663
// MI455X (gfx1250) — matrix operand order repaired and checked by reading, not compiled, not run on hardware
//
#include <hip/hip_runtime.h>
#include <hip/hip_bf16.h>
#include <climits>

typedef __attribute__((ext_vector_type(16))) _Float16 v16h;
typedef __attribute__((ext_vector_type(8)))  _Float16 v8h;
typedef __attribute__((ext_vector_type(8)))  float    v8f;

#define DD 128
#define D3 384
#define NUSER 50000
#define NITEM 30000
#define TOPK 50

// ---------- helpers ----------
__device__ __forceinline__ unsigned f32_ord(float f) {
    unsigned u = __float_as_uint(f);
    return (u & 0x80000000u) ? ~u : (u | 0x80000000u);
}
__device__ __forceinline__ float ord_f32(unsigned u) {
    unsigned v = (u & 0x80000000u) ? (u & 0x7FFFFFFFu) : ~u;
    return __uint_as_float(v);
}
__device__ __forceinline__ float sigmoidf(float x) { return 1.0f / (1.0f + __expf(-x)); }

union frag16 { v16h v; v8h h[2]; };

// A fragment (16-bit A 16x32, ISA 7.12.2): lane L holds row m0+(L&15);
// halves e<8 -> k = kk + (L>>4)*8 + e ; halves e>=8 -> k += 16.
__device__ __forceinline__ v16h load_a_frag(const _Float16* A, int lda, int m0, int kk, int lane) {
    int row = m0 + (lane & 15);
    int hs  = lane >> 4;
    const _Float16* base = A + (size_t)row * lda + kk + hs * 8;
    frag16 f;
    f.h[0] = *(const v8h*)(base);
    f.h[1] = *(const v8h*)(base + 16);
    return f.v;
}
// B fragment (16-bit B 32x16): lane L holds col n0+(L&15); e<8 -> k = kk + (L>>4)*8 + e ; e>=8 -> k += 16.
// BT stored transposed (N x K row-major) so the 16 halves are contiguous.
__device__ __forceinline__ v16h load_b_frag(const _Float16* BT, int ldb, int n0, int kk, int lane) {
    int col = n0 + (lane & 15);
    int ko  = kk + ((lane >> 4) << 3);
    const _Float16* base = BT + (size_t)col * ldb + ko;
    frag16 f;
    f.h[0] = *(const v8h*)(base);
    f.h[1] = *(const v8h*)(base + 16);
    return f.v;
}

// ---------- init ----------
__global__ void k_fill_u32(unsigned* p, long n, unsigned v) {
    long i = (long)blockIdx.x * blockDim.x + threadIdx.x;
    long s = (long)gridDim.x * blockDim.x;
    for (; i < n; i += s) p[i] = v;
}

// ---------- prep ----------
__global__ void k_gi(const float* __restrict__ rel_table, const float* __restrict__ W_i,
                     const float* __restrict__ b_i, float* __restrict__ gi) {
    int r = blockIdx.x, c = threadIdx.x;       // block 384
    float s = b_i[c];
    for (int k = 0; k < DD; ++k) s += rel_table[r * DD + k] * W_i[k * D3 + c];
    gi[r * D3 + c] = s;
}

__global__ void k_convert_hidden(const float* __restrict__ hidden, const int* __restrict__ old_idx,
                                 _Float16* __restrict__ hidden16, _Float16* __restrict__ hp16, int NPREV) {
    int idx = blockIdx.x * blockDim.x + threadIdx.x;
    if (idx >= NPREV * DD) return;
    int j = idx >> 7, d = idx & 127;
    float v = hidden[idx];
    hidden16[idx] = (_Float16)v;
    hp16[(size_t)old_idx[j] * DD + d] = (_Float16)v;   // h_prev_new scatter
}

__global__ void k_weights(const float* __restrict__ W_h, const float* __restrict__ W_pna,
                          _Float16* __restrict__ WhT, _Float16* __restrict__ WcT, _Float16* __restrict__ W4T) {
    int i = blockIdx.x * blockDim.x + threadIdx.x;
    if (i < D3 * DD) { int n = i >> 7, k = i & 127; WhT[i] = (_Float16)W_h[k * D3 + n]; }
    if (i < D3 * 512) { int n = i >> 9, k = i & 511;
        WcT[i] = (_Float16)W_pna[(size_t)((n >> 7) * 512 + k) * DD + (n & 127)]; }
    if (i < DD * DD) { int n = i >> 7, k = i & 127; W4T[i] = (_Float16)W_pna[(size_t)(1536 + k) * DD + n]; }
}

__global__ void k_keep0(const int* __restrict__ old_idx, float* keep0_ws, float* keep_out, int NPREV) {
    int j = blockIdx.x * blockDim.x + threadIdx.x;
    if (j >= NPREV) return;
    int idx = old_idx[j];
    keep0_ws[idx] = 1.0f; keep_out[idx] = 1.0f;
}

// ---------- GEMM 1: gh_table = hidden @ W_h + b_h  (M x 128) x (128 x 384) ----------
__global__ void k_gemm_gh(const _Float16* __restrict__ A, const _Float16* __restrict__ BT,
                          const float* __restrict__ bias, float* __restrict__ C, int M) {
    int lane = threadIdx.x & 31, wave = threadIdx.x >> 5;
    int m0 = blockIdx.x * 16;
    int n0 = (blockIdx.y * 8 + wave) * 16;
    if (m0 + 16 > M || n0 + 16 > D3) return;
    v8f acc = {};
    for (int kk = 0; kk < DD; kk += 32) {
        v16h a = load_a_frag(A, DD, m0, kk, lane);
        v16h b = load_b_frag(BT, DD, n0, kk, lane);
        acc = __builtin_amdgcn_wmma_f32_16x16x32_f16(false, a, false, b, (short)0, acc, false, false);
    }
    int col = n0 + (lane & 15);
    int rbase = m0 + ((lane >> 4) << 3);
    float bv = bias[col];
    for (int v = 0; v < 8; ++v) C[(size_t)(rbase + v) * D3 + col] = acc[v] + bv;
}

// ---------- edge phase: GRU message + segment sum/sumsq/min/max/deg via atomics ----------
__global__ void k_edge(const int* __restrict__ edges, const float* __restrict__ hidden,
                       const float* __restrict__ gi, const float* __restrict__ gh,
                       float* s1, float* s2, unsigned* mn, unsigned* mx, float* deg, int E) {
    int e = blockIdx.x;
    if (e >= E) return;
    int d = threadIdx.x;                              // 128 threads
    int rel = edges[e * 6 + 2];
    int sub = edges[e * 6 + 4];
    int obj = edges[e * 6 + 5];
    size_t gio = (size_t)rel * D3, gho = (size_t)sub * D3;
    float i_r = gi[gio + d], i_z = gi[gio + 128 + d], i_n = gi[gio + 256 + d];
    float h_r = gh[gho + d], h_z = gh[gho + 128 + d], h_n = gh[gho + 256 + d];
    float hs  = hidden[(size_t)sub * DD + d];
    float r = sigmoidf(i_r + h_r);
    float z = sigmoidf(i_z + h_z);
    float n = tanhf(i_n + r * h_n);
    float m = (1.0f - z) * n + z * hs;
    size_t o = (size_t)obj * DD + d;
    atomicAdd(&s1[o], m);
    atomicAdd(&s2[o], m * m);
    atomicMin(&mn[o], f32_ord(m));
    atomicMax(&mx[o], f32_ord(m));
    if (d == 0) atomicAdd(&deg[obj], 1.0f);
}

// ---------- finalize PNA stats -> f16 agg, amp/att scalars ----------
__global__ void k_finalize(const float* __restrict__ s1, const float* __restrict__ s2,
                           const unsigned* __restrict__ mn, const unsigned* __restrict__ mx,
                           const float* __restrict__ deg, _Float16* __restrict__ agg16,
                           float* __restrict__ amp, float* __restrict__ att) {
    int i = blockIdx.x, d = threadIdx.x;              // 128 threads
    float dg = deg[i];
    float dc = fmaxf(dg, 1.0f);
    size_t o = (size_t)i * DD + d;
    float mean = s1[o] / dc;
    float var  = fmaxf(s2[o] / dc - mean * mean, 0.0f);
    float sd   = sqrtf(var + 1e-5f);
    bool  has  = dg > 0.0f;
    float mnv  = has ? ord_f32(mn[o]) : 0.0f;
    float mxv  = has ? ord_f32(mx[o]) : 0.0f;
    size_t a = (size_t)i * 512 + d;
    agg16[a]       = (_Float16)mean;
    agg16[a + 128] = (_Float16)sd;
    agg16[a + 256] = (_Float16)mnv;
    agg16[a + 384] = (_Float16)mxv;
    if (d == 0) {
        float logd = log1pf(dg);     // DELTA = 1
        amp[i] = logd;
        att[i] = 1.0f / (logd + 1.0f);
    }
}

// ---------- GEMM 2: h_tilde = agg@W1 + amp*(agg@W2) + att*(agg@W3) + hp@W4 + b ----------
__global__ void k_gemm_pna(const _Float16* __restrict__ AGG, const _Float16* __restrict__ WcT,
                           const _Float16* __restrict__ HP, const _Float16* __restrict__ W4T,
                           const float* __restrict__ amp, const float* __restrict__ att,
                           const float* __restrict__ bias, float* __restrict__ C, int N) {
    int lane = threadIdx.x & 31, wave = threadIdx.x >> 5;
    int m0 = blockIdx.x * 16;
    int n0 = wave * 16;                                // N-dim = 128 -> 8 waves
    if (m0 + 16 > N) return;
    int rbase = m0 + ((lane >> 4) << 3);
    float sc1[8], sc2[8];
    for (int v = 0; v < 8; ++v) { sc1[v] = amp[rbase + v]; sc2[v] = att[rbase + v]; }
    v8f acc = {};
    for (int seg = 0; seg < 3; ++seg) {
        v8f s = {};
        for (int kk = 0; kk < 512; kk += 32) {
            v16h a = load_a_frag(AGG, 512, m0, kk, lane);
            v16h b = load_b_frag(WcT, 512, seg * 128 + n0, kk, lane);
            s = __builtin_amdgcn_wmma_f32_16x16x32_f16(false, a, false, b, (short)0, s, false, false);
        }
        float* scl = (seg == 1) ? sc1 : sc2;
        if (seg == 0) { for (int v = 0; v < 8; ++v) acc[v] += s[v]; }
        else          { for (int v = 0; v < 8; ++v) acc[v] += scl[v] * s[v]; }
    }
    {   // + h_prev_new @ W4
        v8f s = {};
        for (int kk = 0; kk < DD; kk += 32) {
            v16h a = load_a_frag(HP, DD, m0, kk, lane);
            v16h b = load_b_frag(W4T, DD, n0, kk, lane);
            s = __builtin_amdgcn_wmma_f32_16x16x32_f16(false, a, false, b, (short)0, s, false, false);
        }
        for (int v = 0; v < 8; ++v) acc[v] += s[v];
    }
    int col = n0 + (lane & 15);
    float bv = bias[col];
    for (int v = 0; v < 8; ++v) C[(size_t)(rbase + v) * DD + col] = acc[v] + bv;
}

// ---------- center / scoring ----------
__global__ void k_center_scan(const int* __restrict__ nodes, const int* __restrict__ q_sub,
                              int* center_idx, int N, int B) {
    int i = blockIdx.x * blockDim.x + threadIdx.x;
    if (i >= N) return;
    int b = nodes[i * 2], ent = nodes[i * 2 + 1];
    if (b >= 0 && b < B && ent == q_sub[b]) atomicMin(&center_idx[b], i);
}
__global__ void k_huser(const int* __restrict__ center_idx, const float* __restrict__ h_tilde,
                        float* __restrict__ h_user) {
    int b = blockIdx.x, d = threadIdx.x;
    int ci = center_idx[b];
    if (ci == INT_MAX) ci = 0;                         // argmax of all-false mask -> 0
    h_user[(size_t)b * DD + d] = h_tilde[(size_t)ci * DD + d];
}

__global__ void k_score(const float* __restrict__ h_tilde, const float* __restrict__ h_user,
                        const int* __restrict__ nodes, const float* __restrict__ Wsc,
                        const float* __restrict__ bsc, float* __restrict__ out_hidden,
                        float* __restrict__ out_alpha) {
    int i = blockIdx.x, d = threadIdx.x;               // 128 threads
    __shared__ float red[DD];
    int b = nodes[i * 2];
    float ht = h_tilde[(size_t)i * DD + d];
    red[d] = h_user[(size_t)b * DD + d] * Wsc[d] + ht * Wsc[DD + d];
    __syncthreads();
    for (int s = 64; s > 0; s >>= 1) {
        if (d < s) red[d] += red[d + s];
        __syncthreads();
    }
    float alpha = sigmoidf(red[0] + bsc[0]);           // TAU = 1
    out_hidden[(size_t)i * DD + d] = alpha * ht;
    if (d == 0) out_alpha[i] = alpha;
}

// ---------- top-K per batch (device-side branch on id_layer) ----------
__global__ void k_topk(const float* __restrict__ alpha, const int* __restrict__ nodes,
                       const float* __restrict__ keep0, const int* __restrict__ id_layer,
                       const int* __restrict__ n_layer, float* __restrict__ keep_out, int N) {
    int b = blockIdx.x, t = threadIdx.x;               // 256 threads, B blocks
    int il = id_layer[0], nl = n_layer[0];
    if (il >= nl - 1) {
        for (int i = b * blockDim.x + t; i < N; i += gridDim.x * blockDim.x) {
            int ent = nodes[i * 2 + 1];
            keep_out[i] = (ent >= NUSER && ent < NUSER + NITEM) ? 1.0f : 0.0f;
        }
        return;
    }
    __shared__ float sval[256];
    __shared__ int   sidx[256];
    float lastV = __builtin_inff();
    int   lastI = -1;
    for (int pass = 0; pass < TOPK; ++pass) {
        float bv = -__builtin_inff();
        int   bi = INT_MAX;
        for (int i = t; i < N; i += 256) {
            if (nodes[i * 2] != b) continue;
            if (keep0[i] != 0.0f) continue;            // cand = keep0 == 0
            int ent = nodes[i * 2 + 1];
            float sc = alpha[i] + ((ent >= NUSER && ent < NUSER + NITEM) ? 0.05f : 0.0f);
            // exclude already-selected prefix (val desc, idx asc lexicographic)
            if (sc > lastV || (sc == lastV && i <= lastI)) continue;
            if (sc > bv || (sc == bv && i < bi)) { bv = sc; bi = i; }
        }
        sval[t] = bv; sidx[t] = bi;
        __syncthreads();
        for (int s = 128; s > 0; s >>= 1) {
            if (t < s) {
                if (sval[t + s] > sval[t] || (sval[t + s] == sval[t] && sidx[t + s] < sidx[t])) {
                    sval[t] = sval[t + s]; sidx[t] = sidx[t + s];
                }
            }
            __syncthreads();
        }
        lastV = sval[0]; lastI = sidx[0];
        if (t == 0 && lastI != INT_MAX) keep_out[lastI] = 1.0f;
        __syncthreads();
    }
}

// ---------- launch ----------
extern "C" void kernel_launch(void* const* d_in, const int* in_sizes, int n_in,
                              void* d_out, int out_size, void* d_ws, size_t ws_size,
                              hipStream_t stream) {
    const int B     = in_sizes[0];
    const int NPREV = in_sizes[2] / DD;
    const int E     = in_sizes[3] / 6;
    const int N     = in_sizes[4] / 2;
    const int NRELA = in_sizes[8] / DD;

    const int*   q_sub     = (const int*)  d_in[0];
    const float* hidden    = (const float*)d_in[2];
    const int*   edges     = (const int*)  d_in[3];
    const int*   nodes     = (const int*)  d_in[4];
    const int*   id_layer  = (const int*)  d_in[5];
    const int*   n_layer   = (const int*)  d_in[6];
    const int*   old_idx   = (const int*)  d_in[7];
    const float* rel_table = (const float*)d_in[8];
    const float* W_i       = (const float*)d_in[9];
    const float* W_h       = (const float*)d_in[10];
    const float* b_i       = (const float*)d_in[11];
    const float* b_h       = (const float*)d_in[12];
    const float* W_pna     = (const float*)d_in[13];
    const float* b_pna     = (const float*)d_in[14];
    const float* W_score   = (const float*)d_in[15];
    const float* b_score   = (const float*)d_in[16];

    float* out_hidden = (float*)d_out;
    float* out_alpha  = out_hidden + (size_t)N * DD;
    float* out_keep   = out_alpha + N;

    // workspace carve (256B aligned)
    char* wp = (char*)d_ws;
    auto carve = [&](size_t bytes) -> void* {
        void* r = (void*)wp;
        wp += (bytes + 255) & ~(size_t)255;
        return r;
    };
    float*     gi       = (float*)    carve((size_t)NRELA * D3 * 4);
    float*     gh       = (float*)    carve((size_t)NPREV * D3 * 4);
    _Float16*  hidden16 = (_Float16*) carve((size_t)NPREV * DD * 2);
    _Float16*  hp16     = (_Float16*) carve((size_t)N * DD * 2);
    _Float16*  WhT      = (_Float16*) carve((size_t)D3 * DD * 2);
    _Float16*  WcT      = (_Float16*) carve((size_t)D3 * 512 * 2);
    _Float16*  W4T      = (_Float16*) carve((size_t)DD * DD * 2);
    float*     deg      = (float*)    carve((size_t)N * 4);
    float*     s1       = (float*)    carve((size_t)N * DD * 4);
    float*     s2       = (float*)    carve((size_t)N * DD * 4);
    unsigned*  mn       = (unsigned*) carve((size_t)N * DD * 4);
    unsigned*  mx       = (unsigned*) carve((size_t)N * DD * 4);
    _Float16*  agg16    = (_Float16*) carve((size_t)N * 512 * 2);
    float*     amp      = (float*)    carve((size_t)N * 4);
    float*     att      = (float*)    carve((size_t)N * 4);
    float*     h_tilde  = (float*)    carve((size_t)N * DD * 4);
    float*     keep0    = (float*)    carve((size_t)N * 4);
    int*       center   = (int*)      carve((size_t)B * 4);
    float*     h_user   = (float*)    carve((size_t)B * DD * 4);

    const int FG = 2048;  // grid-stride fill blocks
    k_fill_u32<<<FG, 256, 0, stream>>>((unsigned*)deg,   (long)N,           0u);
    k_fill_u32<<<FG, 256, 0, stream>>>((unsigned*)s1,    (long)N * DD,      0u);
    k_fill_u32<<<FG, 256, 0, stream>>>((unsigned*)s2,    (long)N * DD,      0u);
    k_fill_u32<<<FG, 256, 0, stream>>>(mn,               (long)N * DD,      0xFFFFFFFFu);
    k_fill_u32<<<FG, 256, 0, stream>>>(mx,               (long)N * DD,      0u);
    k_fill_u32<<<FG, 256, 0, stream>>>((unsigned*)hp16,  (long)N * DD / 2,  0u);
    k_fill_u32<<<FG, 256, 0, stream>>>((unsigned*)keep0, (long)N,           0u);
    k_fill_u32<<<FG, 256, 0, stream>>>((unsigned*)out_keep, (long)N,        0u);
    k_fill_u32<<<1,  256, 0, stream>>>((unsigned*)center,(long)B,           0x7FFFFFFFu);

    k_gi<<<NRELA, D3, 0, stream>>>(rel_table, W_i, b_i, gi);
    k_convert_hidden<<<(NPREV * DD + 255) / 256, 256, 0, stream>>>(hidden, old_idx, hidden16, hp16, NPREV);
    k_weights<<<(D3 * 512 + 255) / 256, 256, 0, stream>>>(W_h, W_pna, WhT, WcT, W4T);
    k_keep0<<<(NPREV + 255) / 256, 256, 0, stream>>>(old_idx, keep0, out_keep, NPREV);

    // gh_table = hidden @ W_h + b_h   (WMMA)
    k_gemm_gh<<<dim3(NPREV / 16, 3), 256, 0, stream>>>(hidden16, WhT, b_h, gh, NPREV);

    // per-edge GRU messages + segment stats
    k_edge<<<E, DD, 0, stream>>>(edges, hidden, gi, gh, s1, s2, mn, mx, deg, E);

    k_finalize<<<N, DD, 0, stream>>>(s1, s2, mn, mx, deg, agg16, amp, att);

    // h_tilde (WMMA, 3 scaled segments + h_prev@W4 fused)
    k_gemm_pna<<<N / 16, 256, 0, stream>>>(agg16, WcT, hp16, W4T, amp, att, b_pna, h_tilde, N);

    k_center_scan<<<(N + 255) / 256, 256, 0, stream>>>(nodes, q_sub, center, N, B);
    k_huser<<<B, DD, 0, stream>>>(center, h_tilde, h_user);
    k_score<<<N, DD, 0, stream>>>(h_tilde, h_user, nodes, W_score, b_score, out_hidden, out_alpha);
    k_topk<<<B, 256, 0, stream>>>(out_alpha, nodes, keep0, id_layer, n_layer, out_keep, N);
}
